// WindowAttention_45543833207268
// MI455X (gfx1250) — compile-verified, matrix operand order checked by reading, not run on hardware
//
#include <hip/hip_runtime.h>
#include <hip/hip_bf16.h>
#include <math.h>

// ---------------------------------------------------------------------------
// Window attention, fused per 64-token window, bf16 WMMA (CDNA5 / gfx1250).
//   qx:(8192,64,192) kvx:(4096,64,192) pos:(1,64,192)
//   q = (qx+pos)@Wq^T+bq ; kv = kvx@Wkv^T+bkv ; per-head(6, hd=32) l2norm q,k
//   attn = softmax(q̂·k̂ / sqrt(192)) ; x = attn@v ; out = x@Wp^T+bp
// Memory-bound-ish (~1GB traffic -> 43us @23.3TB/s; ~143 GFLOP), so fuse
// everything into one pass over HBM and run all GEMMs on v_wmma bf16.
// ---------------------------------------------------------------------------

#define DIM    192
#define WIN    64
#define HEADS  6
#define HD     32

typedef __attribute__((ext_vector_type(16))) __bf16        v16bf;
typedef __attribute__((ext_vector_type(8)))  float         v8f;
typedef __attribute__((ext_vector_type(4)))  unsigned int  u32x4;

union FragU { u32x4 q[2]; v16bf v; };

__device__ __forceinline__ unsigned short f2bf(float f) {
  unsigned int u = __float_as_uint(f);
  u += 0x7FFFu + ((u >> 16) & 1u);          // round-to-nearest-even
  return (unsigned short)(u >> 16);
}

// A-style fragment load from LDS, row-major bf16 [rows][stride].
// 16-bit A layout (16x32): lane L holds row row0+(L&15); K groups:
// lanes 0-15 -> K 0..7 & 16..23 ; lanes 16-31 -> K 8..15 & 24..31.
// Two 16B ds loads per fragment. Also used for B when memory holds B^T rows.
__device__ __forceinline__ v16bf load_frag_lds(const unsigned short* base,
                                               int row0, int stride, int col0,
                                               int lane) {
  int r = row0 + (lane & 15);
  int half = lane >> 4;
  const unsigned short* p = base + r * stride + col0 + half * 8;
  FragU f;
  f.q[0] = *(const u32x4*)(p);        // K = col0 + half*8 + [0..7]
  f.q[1] = *(const u32x4*)(p + 16);   // K = col0 + 16 + half*8 + [0..7]
  return f.v;
}

// B fragment from pre-swizzled global weights: each 16x32k tile stored as
// 32 lanes x 16 bf16 contiguous (lane-major), so a lane reads 32 contiguous B.
__device__ __forceinline__ v16bf load_frag_glb(const unsigned short* sw,
                                               int tile, int lane) {
  const unsigned short* p = sw + ((size_t)tile << 9) + (lane << 4);
  FragU f;
  f.q[0] = *(const u32x4*)(p);
  f.q[1] = *(const u32x4*)(p + 8);
  return f.v;
}

#define WMMA_BF16(A, B, C) \
  __builtin_amdgcn_wmma_f32_16x16x32_bf16(false, (A), false, (B), (short)0, (C), false, false)

// ---------------------------------------------------------------------------
// Kernel 0: swizzle a weight matrix W[N][K] (f32, row-major, y = x@W^T) into
// bf16 B-fragment tiles. tile index = nt*(K/32)+kt ; element j of lane l:
//   n = nt*16 + (l&15);  k = kt*32 + (j/2 /4)*16 + (l>>4)*8 + ((j/2)&3)*2 + (j&1)
// ---------------------------------------------------------------------------
__global__ void wa_swizzle_w(const float* __restrict__ W,
                             unsigned short* __restrict__ out, int N, int K) {
  int KT = K >> 5;
  int tile = blockIdx.x;
  int nt = tile / KT, kt = tile % KT;
  int lane = threadIdx.x;
  int n = nt * 16 + (lane & 15);
  int half = lane >> 4;
  unsigned short* dst = out + ((size_t)tile << 9) + (lane << 4);
#pragma unroll
  for (int j = 0; j < 16; ++j) {
    int i = j >> 1, sub = j & 1;
    int koff = ((i >> 2) << 4) + half * 8 + ((i & 3) << 1) + sub;
    dst[j] = f2bf(W[(size_t)n * K + kt * 32 + koff]);
  }
}

// ---------------------------------------------------------------------------
// Fused window-attention kernel: 1 workgroup (8 wave32) per window.
// LDS regions (byte offsets, all 16B aligned; padded strides vs. 64 banks):
//   sX  [64][200] bf16  @      0  (qx+pos)         25600
//   sKV [64][200] bf16  @  25600  (kvx)            25600
//   sQ  [64][200] bf16  @  51200  (q̂)              25600
//   sK  [64][200] bf16  @  76800  (k̂)              25600
//   sVt [192][72] bf16  @ 102400  (v transposed)   27648
//   sF  f32 scratch     @ 130048                  101376
//       kv-out [64][388] -> q-out [64][196] -> scores [6][64][66]
//   sP  probs [6][64][72] bf16  aliases sX..sQ (they are dead by then)
//   sO  attn-out [64][200] bf16 aliases sF (scores dead by then)
// total 231424 B (dynamic LDS; WGP pool is 320KB)
// ---------------------------------------------------------------------------
#define OFF_X    0
#define OFF_KV   25600
#define OFF_Q    51200
#define OFF_K    76800
#define OFF_VT   102400
#define OFF_F    130048
#define LDS_TOTAL 231424
#define XS   200   // bf16 row stride for 64x192 tensors
#define VTS  72    // bf16 row stride for Vt / probs

__global__ void __launch_bounds__(256)
wa_fused(const float* __restrict__ qx, const float* __restrict__ kvx,
         const float* __restrict__ pos,
         const unsigned short* __restrict__ WqSw, const float* __restrict__ bq,
         const unsigned short* __restrict__ WkvSw, const float* __restrict__ bkv,
         const unsigned short* __restrict__ WpSw, const float* __restrict__ bp,
         float* __restrict__ out, int Bkv) {
  extern __shared__ unsigned char lds[];
  unsigned short* sX  = (unsigned short*)(lds + OFF_X);
  unsigned short* sKV = (unsigned short*)(lds + OFF_KV);
  unsigned short* sQ  = (unsigned short*)(lds + OFF_Q);
  unsigned short* sK  = (unsigned short*)(lds + OFF_K);
  unsigned short* sVt = (unsigned short*)(lds + OFF_VT);
  float*          sF  = (float*)(lds + OFF_F);
  unsigned short* sP  = (unsigned short*)(lds + OFF_X);  // alias (stage 7+)
  unsigned short* sO  = (unsigned short*)(lds + OFF_F);  // alias (stage 8+)

  const int tid  = threadIdx.x;
  const int wave = tid >> 5;
  const int lane = tid & 31;
  const int b    = blockIdx.x;
  const int bk   = b % Bkv;
  const float* qxb = qx + (size_t)b  * (WIN * DIM);
  const float* kvb = kvx + (size_t)bk * (WIN * DIM);

  // --- 1. stage inputs as bf16 -------------------------------------------
  for (int e = tid; e < WIN * DIM; e += 256) {
    int m = e / DIM, c = e % DIM;
    sX[m * XS + c]  = f2bf(qxb[e] + pos[e]);
    sKV[m * XS + c] = f2bf(kvb[e]);
  }
  __syncthreads();

  // --- 2. kv = kvx @ Wkv^T  (M=64,N=384,K=192) -> sF[64][388] ------------
  for (int t = wave; t < 96; t += 8) {
    int mt = t & 3, nt = t >> 2;
    v8f acc = {};
#pragma unroll
    for (int kt = 0; kt < 6; ++kt) {
      v16bf a  = load_frag_lds(sKV, mt * 16, XS, kt * 32, lane);
      v16bf bb = load_frag_glb(WkvSw, nt * 6 + kt, lane);
      acc = WMMA_BF16(a, bb, acc);
    }
    int m0 = mt * 16 + (lane >> 4) * 8;
    int n  = nt * 16 + (lane & 15);
#pragma unroll
    for (int r = 0; r < 8; ++r) sF[(m0 + r) * 388 + n] = acc[r];
  }
  __syncthreads();

  // --- 3. +bkv ; l2norm(k) -> sK ; v -> sVt (transposed) -----------------
  for (int idx = tid; idx < WIN * HEADS; idx += 256) {
    int m = idx / HEADS, h = idx % HEADS;
    const float* kr = sF + m * 388 + h * HD;
    const float* vr = sF + m * 388 + DIM + h * HD;
    float s = 0.f;
    for (int d = 0; d < HD; ++d) { float x = kr[d] + bkv[h * HD + d]; s += x * x; }
    float inv = 1.f / fmaxf(sqrtf(s), 1e-12f);
    for (int d = 0; d < HD; ++d) {
      float x = kr[d] + bkv[h * HD + d];
      sK[m * XS + h * HD + d] = f2bf(x * inv);
      float v = vr[d] + bkv[DIM + h * HD + d];
      sVt[(h * HD + d) * VTS + m] = f2bf(v);
    }
  }
  __syncthreads();

  // --- 4. q = (qx+pos) @ Wq^T  (M=64,N=192,K=192) -> sF[64][196] ---------
  for (int t = wave; t < 48; t += 8) {
    int mt = t & 3, nt = t >> 2;
    v8f acc = {};
#pragma unroll
    for (int kt = 0; kt < 6; ++kt) {
      v16bf a  = load_frag_lds(sX, mt * 16, XS, kt * 32, lane);
      v16bf bb = load_frag_glb(WqSw, nt * 6 + kt, lane);
      acc = WMMA_BF16(a, bb, acc);
    }
    int m0 = mt * 16 + (lane >> 4) * 8;
    int n  = nt * 16 + (lane & 15);
#pragma unroll
    for (int r = 0; r < 8; ++r) sF[(m0 + r) * 196 + n] = acc[r];
  }
  __syncthreads();

  // --- 5. +bq ; l2norm(q) -> sQ ------------------------------------------
  for (int idx = tid; idx < WIN * HEADS; idx += 256) {
    int m = idx / HEADS, h = idx % HEADS;
    const float* qr = sF + m * 196 + h * HD;
    float s = 0.f;
    for (int d = 0; d < HD; ++d) { float x = qr[d] + bq[h * HD + d]; s += x * x; }
    float inv = 1.f / fmaxf(sqrtf(s), 1e-12f);
    for (int d = 0; d < HD; ++d) {
      float x = qr[d] + bq[h * HD + d];
      sQ[m * XS + h * HD + d] = f2bf(x * inv);
    }
  }
  __syncthreads();

  // --- 6. scores = q̂ k̂^T * scale  -> sF[h][64][66]  (K=HD=32: 1 wmma) ---
  const float scale = 0.07216878364870323f;  // 1/sqrt(192)
  for (int t = wave; t < 96; t += 8) {
    int h = t >> 4, r2 = t & 15, mt = r2 >> 2, nt = r2 & 3;
    v16bf a  = load_frag_lds(sQ, mt * 16, XS, h * HD, lane);
    v16bf bb = load_frag_lds(sK, nt * 16, XS, h * HD, lane);  // B = k̂^T rows
    v8f acc = {};
    acc = WMMA_BF16(a, bb, acc);
    int m0 = mt * 16 + (lane >> 4) * 8;
    int n  = nt * 16 + (lane & 15);
#pragma unroll
    for (int r = 0; r < 8; ++r) sF[h * 4224 + (m0 + r) * 66 + n] = acc[r] * scale;
  }
  __syncthreads();

  // --- 7. softmax rows -> probs bf16 sP[h][64][72] (aliases sX..sQ) ------
  for (int idx = tid; idx < HEADS * WIN; idx += 256) {
    int h = idx >> 6, m = idx & 63;
    const float* row = sF + h * 4224 + m * 66;
    float mx = row[0];
    for (int n = 1; n < WIN; ++n) mx = fmaxf(mx, row[n]);
    float sum = 0.f;
    for (int n = 0; n < WIN; ++n) sum += __expf(row[n] - mx);
    float inv = 1.f / sum;
    unsigned short* pr = sP + (h * WIN + m) * VTS;
    for (int n = 0; n < WIN; ++n) pr[n] = f2bf(__expf(row[n] - mx) * inv);
  }
  __syncthreads();

  // --- 8. x = attn @ v  (per head M=64,N=32,K=64) -> sO (aliases sF) -----
  for (int t = wave; t < 48; t += 8) {
    int h = t / 8, r2 = t % 8, mt = r2 >> 1, nt = r2 & 1;
    v8f acc = {};
#pragma unroll
    for (int kt = 0; kt < 2; ++kt) {
      v16bf a  = load_frag_lds(sP + (size_t)h * WIN * VTS, mt * 16, VTS, kt * 32, lane);
      v16bf bb = load_frag_lds(sVt, h * HD + nt * 16, VTS, kt * 32, lane);  // B = V via Vt rows
      acc = WMMA_BF16(a, bb, acc);
    }
    int m0 = mt * 16 + (lane >> 4) * 8;
    int n  = h * HD + nt * 16 + (lane & 15);
#pragma unroll
    for (int r = 0; r < 8; ++r) sO[(m0 + r) * XS + n] = f2bf(acc[r]);
  }
  __syncthreads();

  // --- 9. out = x @ Wp^T + bp -> global (f32) ----------------------------
  float* ob = out + (size_t)b * (WIN * DIM);
  for (int t = wave; t < 48; t += 8) {
    int mt = t & 3, nt = t >> 2;
    v8f acc = {};
#pragma unroll
    for (int kt = 0; kt < 6; ++kt) {
      v16bf a  = load_frag_lds(sO, mt * 16, XS, kt * 32, lane);
      v16bf bb = load_frag_glb(WpSw, nt * 6 + kt, lane);
      acc = WMMA_BF16(a, bb, acc);
    }
    int n = nt * 16 + (lane & 15);
    float bias = bp[n];
    int m0 = mt * 16 + (lane >> 4) * 8;
#pragma unroll
    for (int r = 0; r < 8; ++r) ob[(size_t)(m0 + r) * DIM + n] = acc[r] + bias;
  }
}

// ---------------------------------------------------------------------------
extern "C" void kernel_launch(void* const* d_in, const int* in_sizes, int n_in,
                              void* d_out, int out_size, void* d_ws, size_t ws_size,
                              hipStream_t stream) {
  const float* qx  = (const float*)d_in[0];
  const float* kvx = (const float*)d_in[1];
  const float* pos = (const float*)d_in[2];
  const float* Wq  = (const float*)d_in[3];
  const float* bq  = (const float*)d_in[4];
  const float* Wkv = (const float*)d_in[5];
  const float* bkv = (const float*)d_in[6];
  const float* Wp  = (const float*)d_in[7];
  const float* bp  = (const float*)d_in[8];
  float* out = (float*)d_out;

  int Bq  = in_sizes[0] / (WIN * DIM);   // 8192
  int Bkv = in_sizes[1] / (WIN * DIM);   // 4096

  unsigned short* wsQ  = (unsigned short*)d_ws;            // 192x192 bf16
  unsigned short* wsKV = wsQ + 192 * 192;                  // 384x192 bf16
  unsigned short* wsP  = wsKV + 384 * 192;                 // 192x192 bf16

  // Pre-swizzle weights into WMMA B-fragment tiles (bf16), reused by all 8192 WGs.
  wa_swizzle_w<<<(192 / 16) * (192 / 32), 32, 0, stream>>>(Wq, wsQ, 192, 192);
  wa_swizzle_w<<<(384 / 16) * (192 / 32), 32, 0, stream>>>(Wkv, wsKV, 384, 192);
  wa_swizzle_w<<<(192 / 16) * (192 / 32), 32, 0, stream>>>(Wp, wsP, 192, 192);

  wa_fused<<<Bq, 256, LDS_TOTAL, stream>>>(qx, kvx, pos, wsQ, bq, wsKV, bkv,
                                           wsP, bp, out, Bkv);
}
